// SpatioTemporalTransformerGNN_15023795602154
// MI455X (gfx1250) — matrix operand order repaired and checked by reading, not compiled, not run on hardware
//
#include <hip/hip_runtime.h>
#include <hip/hip_bf16.h>

#define DEVINL __device__ __forceinline__

typedef __attribute__((ext_vector_type(16))) _Float16 v16h;
typedef __attribute__((ext_vector_type(8)))  _Float16 v8h;
typedef __attribute__((ext_vector_type(8)))  float    v8f;

namespace {
constexpr int   N_     = 20000;
constexpr int   E_     = 320000;
constexpr int   EN_    = E_ + N_;       // edges + self loops
constexpr int   T_     = 6;
constexpr int   FIN_   = 64;
constexpr int   HID_   = 128;
constexpr int   HEADS_ = 4;
constexpr int   HC_    = 512;           // HEADS * HID
constexpr int   OUT_   = 3;
constexpr float NEG_   = 0.2f;
constexpr float EPS_   = 1e-5f;
}

// ---------------- device helpers ----------------

DEVINL float wsum(float v) {
  v += __shfl_xor(v, 1, 32);
  v += __shfl_xor(v, 2, 32);
  v += __shfl_xor(v, 4, 32);
  v += __shfl_xor(v, 8, 32);
  v += __shfl_xor(v, 16, 32);
  return v;
}
DEVINL unsigned fenc(float f) {          // order-preserving float->uint
  unsigned u = __float_as_uint(f);
  return (u & 0x80000000u) ? ~u : (u | 0x80000000u);
}
DEVINL float fdec(unsigned e) {
  unsigned u = (e & 0x80000000u) ? (e ^ 0x80000000u) : ~e;
  return __uint_as_float(u);
}
DEVINL float lrelu(float x) { return x > 0.f ? x : NEG_ * x; }
DEVINL float eluf(float x)  { return x > 0.f ? x : __expf(x) - 1.f; }
DEVINL int getsrc(const int* ei, int e) { return (e < E_) ? ei[e]      : (e - E_); }
DEVINL int getdst(const int* ei, int e) { return (e < E_) ? ei[E_ + e] : (e - E_); }

// ---------------- small utility kernels ----------------

__global__ void fill_k(unsigned* p, int count, unsigned val) {
  int i = blockIdx.x * blockDim.x + threadIdx.x;
  if (i < count) p[i] = val;
}

__global__ void f2h_k(const float* __restrict__ src, _Float16* __restrict__ dst, int count) {
  int i = blockIdx.x * blockDim.x + threadIdx.x;
  if (i < count) dst[i] = (_Float16)src[i];
}

__global__ void edge_sum_k(const float* __restrict__ a, float* __restrict__ s) {
  __shared__ float sm[256];
  float v = 0.f;
  for (int i = blockIdx.x * blockDim.x + threadIdx.x; i < E_; i += gridDim.x * blockDim.x)
    v += a[i];
  sm[threadIdx.x] = v;
  __syncthreads();
  for (int st = 128; st > 0; st >>= 1) {
    if (threadIdx.x < st) sm[threadIdx.x] += sm[threadIdx.x + st];
    __syncthreads();
  }
  if (threadIdx.x == 0) atomicAdd(s, sm[0]);
}

__global__ void ea_fill_k(const float* __restrict__ eattr, const float* __restrict__ fsum,
                          float* __restrict__ ea) {
  int i = blockIdx.x * blockDim.x + threadIdx.x;
  if (i >= EN_) return;
  ea[i] = (i < E_) ? eattr[i] : fsum[0] * (1.0f / (float)E_);
}

__global__ void conv_xt_k(const float* __restrict__ x, _Float16* __restrict__ xt16, int t) {
  int i = blockIdx.x * blockDim.x + threadIdx.x;
  if (i >= N_ * FIN_) return;
  int n = i >> 6, c = i & 63;
  xt16[i] = (_Float16)x[((size_t)n * T_ + t) * FIN_ + c];
}

// h = elu(acc + bias) -> f16 (input to next GEMM)
__global__ void elu_h16_k(const float* __restrict__ acc, const float* __restrict__ bias,
                          _Float16* __restrict__ h16) {
  int i = blockIdx.x * blockDim.x + threadIdx.x;
  if (i >= N_ * HC_) return;
  int c = i & (HC_ - 1);
  h16[i] = (_Float16)eluf(acc[i] + bias[c]);
}

// seq[:, t, :] = elu(acc2 + bias2)
__global__ void seq_write_k(const float* __restrict__ acc, const float* __restrict__ bias,
                            float* __restrict__ seq, int t) {
  int i = blockIdx.x * blockDim.x + threadIdx.x;
  if (i >= N_ * HID_) return;
  int n = i >> 7, c = i & 127;
  seq[((size_t)n * T_ + t) * HID_ + c] = eluf(acc[i] + bias[c]);
}

// ---------------- WMMA GEMM:  C[M,Nout] = A[M,K] * W[Nout,K]^T + bias ----------------
// One wave computes a 16x64 strip: 4 accumulator tiles sharing one A fragment per
// K-step (v_wmma_f32_16x16x32_f16). 40 B/lane of loads per WMMA instead of 64 B.
// A fragment (16-bit 16x32): lanes 0-15 hold row M=l, K [k0, k0+16);
//                            lanes 16-31 hold the same rows, K [k0+16, k0+32).
// B fragment (16-bit 32x16): lanes 0-15 hold col N=l, K [k0, k0+16); lanes 16-31 K [k0+16, k0+32).
__global__ void gemm_wmma_k(const _Float16* __restrict__ A, const _Float16* __restrict__ W,
                            const float* __restrict__ bias, float* __restrict__ C,
                            int M, int K, int Nout) {
  constexpr int NT = 4;                 // column tiles per wave (64 columns)
  int gw   = (blockIdx.x * blockDim.x + threadIdx.x) >> 5;
  int lane = threadIdx.x & 31;
  int tilesNG = Nout >> 6;              // groups of 64 columns
  int tilesM  = (M + 15) >> 4;
  if (gw >= tilesM * tilesNG) return;   // wave-uniform: EXEC stays all-1s
  int tm = gw / tilesNG, tg = gw % tilesNG;
  int m0 = tm << 4, n0 = tg << 6;
  int half = lane >> 4;
  int l    = lane & 15;
  const _Float16* Arow = A + (size_t)(m0 + l) * K;
  const _Float16* Wrow[NT];
#pragma unroll
  for (int j = 0; j < NT; ++j)
    Wrow[j] = W + (size_t)(n0 + j * 16 + l) * K + half * 16;
  v8f acc[NT] = {};
  for (int k0 = 0; k0 < K; k0 += 32) {
    int ka = k0 + half * 16;
    v16h a = *(const v16h*)(Arow + ka);
    __builtin_prefetch(Arow + ka + 64, 0, 1);      // next K-chunk -> global_prefetch_b8
#pragma unroll
    for (int j = 0; j < NT; ++j) {
      v16h b = *(const v16h*)(Wrow[j] + k0);
      acc[j] = __builtin_amdgcn_wmma_f32_16x16x32_f16(false, a, false, b, (short)0, acc[j],
                                                      false, false);
    }
  }
#pragma unroll
  for (int j = 0; j < NT; ++j) {
    int n = n0 + j * 16 + l;
    float bv = bias[n];
#pragma unroll
    for (int r = 0; r < 8; ++r) {
      int m = m0 + r + half * 8;                   // C layout: VGPR r -> M=r / M=r+8
      if (m < M) C[(size_t)m * Nout + n] = acc[j][r] + bv;
    }
  }
}

// ---------------- GAT layer 1 edge kernels (H=4, C=128) ----------------

__global__ void gat1_logits_k(const float* __restrict__ xl, const float* __restrict__ xr,
                              const float* __restrict__ ea, const int* __restrict__ ei,
                              const float* __restrict__ We, const float* __restrict__ att,
                              float* __restrict__ lg, unsigned* __restrict__ mx) {
  int gw = (blockIdx.x * blockDim.x + threadIdx.x) >> 5;
  int lane = threadIdx.x & 31;
  if (gw >= EN_) return;
  int s = getsrc(ei, gw), d = getdst(ei, gw);
  float eav = ea[gw];
  const float* xls = xl + (size_t)s * HC_;
  const float* xrd = xr + (size_t)d * HC_;
#pragma unroll
  for (int h = 0; h < HEADS_; ++h) {
    int base = h * HID_ + lane * 4;
    float4 a = *(const float4*)(xls + base);
    float4 b = *(const float4*)(xrd + base);
    float4 w = *(const float4*)(We + base);
    float4 t = *(const float4*)(att + base);
    float p = lrelu(a.x + b.x + eav * w.x) * t.x
            + lrelu(a.y + b.y + eav * w.y) * t.y
            + lrelu(a.z + b.z + eav * w.z) * t.z
            + lrelu(a.w + b.w + eav * w.w) * t.w;
    float logit = wsum(p);
    if (lane == 0) {
      lg[(size_t)gw * HEADS_ + h] = logit;
      atomicMax(mx + (size_t)d * HEADS_ + h, fenc(logit));
    }
  }
}

__global__ void gat1_soft_k(const int* __restrict__ ei, const unsigned* __restrict__ mx,
                            float* __restrict__ lg, float* __restrict__ den) {
  int i = blockIdx.x * blockDim.x + threadIdx.x;
  if (i >= EN_ * HEADS_) return;
  int e = i >> 2, h = i & 3;
  int d = getdst(ei, e);
  float m = fdec(mx[(size_t)d * HEADS_ + h]);
  float ex = __expf(lg[i] - m);
  lg[i] = ex;
  atomicAdd(den + (size_t)d * HEADS_ + h, ex);
}

__global__ void gat1_aggr_k(const float* __restrict__ xl, const int* __restrict__ ei,
                            const float* __restrict__ lg, const float* __restrict__ den,
                            float* __restrict__ acc) {
  int gw = (blockIdx.x * blockDim.x + threadIdx.x) >> 5;
  int lane = threadIdx.x & 31;
  if (gw >= EN_) return;
  int s = getsrc(ei, gw), d = getdst(ei, gw);
  const float* xls = xl + (size_t)s * HC_;
  float* ad = acc + (size_t)d * HC_;
#pragma unroll
  for (int h = 0; h < HEADS_; ++h) {
    float alpha = lg[(size_t)gw * HEADS_ + h] / den[(size_t)d * HEADS_ + h];
    int base = h * HID_ + lane * 4;
    float4 a = *(const float4*)(xls + base);
    atomicAdd(ad + base + 0, alpha * a.x);
    atomicAdd(ad + base + 1, alpha * a.y);
    atomicAdd(ad + base + 2, alpha * a.z);
    atomicAdd(ad + base + 3, alpha * a.w);
  }
}

// ---------------- GAT layer 2 edge kernels (H=1, C=128) ----------------

__global__ void gat2_logits_k(const float* __restrict__ xl, const float* __restrict__ xr,
                              const float* __restrict__ ea, const int* __restrict__ ei,
                              const float* __restrict__ We, const float* __restrict__ att,
                              float* __restrict__ lg, unsigned* __restrict__ mx) {
  int gw = (blockIdx.x * blockDim.x + threadIdx.x) >> 5;
  int lane = threadIdx.x & 31;
  if (gw >= EN_) return;
  int s = getsrc(ei, gw), d = getdst(ei, gw);
  float eav = ea[gw];
  int base = lane * 4;
  float4 a = *(const float4*)(xl + (size_t)s * HID_ + base);
  float4 b = *(const float4*)(xr + (size_t)d * HID_ + base);
  float4 w = *(const float4*)(We + base);
  float4 t = *(const float4*)(att + base);
  float p = lrelu(a.x + b.x + eav * w.x) * t.x
          + lrelu(a.y + b.y + eav * w.y) * t.y
          + lrelu(a.z + b.z + eav * w.z) * t.z
          + lrelu(a.w + b.w + eav * w.w) * t.w;
  float logit = wsum(p);
  if (lane == 0) {
    lg[gw] = logit;
    atomicMax(mx + d, fenc(logit));
  }
}

__global__ void gat2_soft_k(const int* __restrict__ ei, const unsigned* __restrict__ mx,
                            float* __restrict__ lg, float* __restrict__ den) {
  int e = blockIdx.x * blockDim.x + threadIdx.x;
  if (e >= EN_) return;
  int d = getdst(ei, e);
  float m = fdec(mx[d]);
  float ex = __expf(lg[e] - m);
  lg[e] = ex;
  atomicAdd(den + d, ex);
}

__global__ void gat2_aggr_k(const float* __restrict__ xl, const int* __restrict__ ei,
                            const float* __restrict__ lg, const float* __restrict__ den,
                            float* __restrict__ acc) {
  int gw = (blockIdx.x * blockDim.x + threadIdx.x) >> 5;
  int lane = threadIdx.x & 31;
  if (gw >= EN_) return;
  int s = getsrc(ei, gw), d = getdst(ei, gw);
  float alpha = lg[gw] / den[d];
  int base = lane * 4;
  float4 a = *(const float4*)(xl + (size_t)s * HID_ + base);
  float* ad = acc + (size_t)d * HID_ + base;
  atomicAdd(ad + 0, alpha * a.x);
  atomicAdd(ad + 1, alpha * a.y);
  atomicAdd(ad + 2, alpha * a.z);
  atomicAdd(ad + 3, alpha * a.w);
}

// ---------------- fused temporal attention + LN + MLP head ----------------
// One wave per node. Only y[:, T-1, :] is needed downstream, so q is computed
// only for the last step while k/v cover all T. Channel c = 32*j + lane maps
// slot j -> head j, lane -> d within head, so scores are wave reductions and
// the weighted sum over v is lane-local.
__global__ void __launch_bounds__(256) temporal_k(
    const float* __restrict__ seq,
    const float* __restrict__ Wqkv, const float* __restrict__ bqkv,
    const float* __restrict__ Wo,   const float* __restrict__ bo,
    const float* __restrict__ lng,  const float* __restrict__ lnb,
    const float* __restrict__ Wf1,  const float* __restrict__ bf1,
    const float* __restrict__ Wout, const float* __restrict__ bout,
    float* __restrict__ out) {
  __shared__ float srow[8][T_ * HID_];
  __shared__ float obuf[8][HID_];
  __shared__ float ybuf[8][HID_];
  __shared__ float hbuf[8][64];
  int wid = threadIdx.x >> 5, lane = threadIdx.x & 31;
  int node = blockIdx.x * 8 + wid;
  bool active = node < N_;
  int n = active ? node : (N_ - 1);
  const float* sp = seq + (size_t)n * T_ * HID_;
  for (int i = lane; i < (T_ * HID_) / 4; i += 32)
    ((float4*)srow[wid])[i] = ((const float4*)sp)[i];
  __syncthreads();

  float kreg[T_][HEADS_], vreg[T_][HEADS_], qreg[HEADS_];
  for (int t = 0; t < T_; ++t) {
#pragma unroll
    for (int j = 0; j < HEADS_; ++j) {
      int c = 32 * j + lane;
      const float* wk = Wqkv + (size_t)(HID_ + c) * HID_;
      const float* wv = Wqkv + (size_t)(2 * HID_ + c) * HID_;
      float ak = bqkv[HID_ + c], av = bqkv[2 * HID_ + c];
      const float* s = srow[wid] + t * HID_;
      for (int i = 0; i < HID_; ++i) { float sv = s[i]; ak += wk[i] * sv; av += wv[i] * sv; }
      kreg[t][j] = ak; vreg[t][j] = av;
    }
  }
#pragma unroll
  for (int j = 0; j < HEADS_; ++j) {
    int c = 32 * j + lane;
    const float* wq = Wqkv + (size_t)c * HID_;
    float aq = bqkv[c];
    const float* s = srow[wid] + (T_ - 1) * HID_;
    for (int i = 0; i < HID_; ++i) aq += wq[i] * s[i];
    qreg[j] = aq;
  }

  const float scale = 0.17677669529663687f;  // 1/sqrt(32)
#pragma unroll
  for (int j = 0; j < HEADS_; ++j) {
    float sc[T_];
    float mx = -3.4e38f;
    for (int t = 0; t < T_; ++t) {
      sc[t] = wsum(qreg[j] * kreg[t][j]) * scale;
      mx = fmaxf(mx, sc[t]);
    }
    float ssum = 0.f;
    for (int t = 0; t < T_; ++t) { sc[t] = __expf(sc[t] - mx); ssum += sc[t]; }
    float inv = 1.f / ssum;
    float ov = 0.f;
    for (int t = 0; t < T_; ++t) ov += sc[t] * inv * vreg[t][j];
    obuf[wid][32 * j + lane] = ov;
  }
  __syncthreads();

  float yv[HEADS_], ps = 0.f, pq = 0.f;
#pragma unroll
  for (int j = 0; j < HEADS_; ++j) {
    int c = 32 * j + lane;
    float a = bo[c] + srow[wid][(T_ - 1) * HID_ + c];
    const float* wr = Wo + (size_t)c * HID_;
    for (int i = 0; i < HID_; ++i) a += wr[i] * obuf[wid][i];
    yv[j] = a; ps += a; pq += a * a;
  }
  float mu   = wsum(ps) * (1.f / HID_);
  float var  = wsum(pq) * (1.f / HID_) - mu * mu;
  float rstd = rsqrtf(var + EPS_);
#pragma unroll
  for (int j = 0; j < HEADS_; ++j) {
    int c = 32 * j + lane;
    ybuf[wid][c] = (yv[j] - mu) * rstd * lng[c] + lnb[c];
  }
  __syncthreads();
#pragma unroll
  for (int r = 0; r < 2; ++r) {
    int c = 32 * r + lane;
    float a = bf1[c];
    const float* wr = Wf1 + (size_t)c * HID_;
    for (int i = 0; i < HID_; ++i) a += wr[i] * ybuf[wid][i];
    hbuf[wid][c] = fmaxf(a, 0.f);
  }
  __syncthreads();
  if (active && lane < OUT_) {
    float a = bout[lane];
    const float* wr = Wout + (size_t)lane * 64;
    for (int i = 0; i < 64; ++i) a += wr[i] * hbuf[wid][i];
    out[(size_t)node * OUT_ + lane] = a;
  }
}

// ---------------- host launcher ----------------

extern "C" void kernel_launch(void* const* d_in, const int* in_sizes, int n_in,
                              void* d_out, int out_size, void* d_ws, size_t ws_size,
                              hipStream_t stream) {
  (void)in_sizes; (void)n_in; (void)out_size; (void)ws_size;

  const float* x     = (const float*)d_in[0];
  const int*   ei    = (const int*)d_in[1];
  const float* eattr = (const float*)d_in[2];
  const float* Wl1   = (const float*)d_in[3];
  const float* bl1   = (const float*)d_in[4];
  const float* Wr1   = (const float*)d_in[5];
  const float* br1   = (const float*)d_in[6];
  const float* We1   = (const float*)d_in[7];
  const float* att1  = (const float*)d_in[8];
  const float* bias1 = (const float*)d_in[9];
  const float* Wl2   = (const float*)d_in[10];
  const float* bl2   = (const float*)d_in[11];
  const float* Wr2   = (const float*)d_in[12];
  const float* br2   = (const float*)d_in[13];
  const float* We2   = (const float*)d_in[14];
  const float* att2  = (const float*)d_in[15];
  const float* bias2 = (const float*)d_in[16];
  const float* Wqkv  = (const float*)d_in[17];
  const float* bqkv  = (const float*)d_in[18];
  const float* Wo    = (const float*)d_in[19];
  const float* bo    = (const float*)d_in[20];
  const float* ln_g  = (const float*)d_in[21];
  const float* ln_b  = (const float*)d_in[22];
  const float* Wf1   = (const float*)d_in[23];
  const float* bf1   = (const float*)d_in[24];
  const float* Wout  = (const float*)d_in[25];
  const float* bout  = (const float*)d_in[26];
  float* out = (float*)d_out;

  char* ws = (char*)d_ws;
  size_t off = 0;
  auto carve = [&](size_t bytes) -> void* {
    void* p = ws + off;
    off += (bytes + 255) & ~(size_t)255;
    return p;
  };

  float*    fsum  = (float*)carve(4);
  float*    ea    = (float*)carve((size_t)EN_ * 4);
  _Float16* Wl1h  = (_Float16*)carve((size_t)HC_ * FIN_ * 2);
  _Float16* Wr1h  = (_Float16*)carve((size_t)HC_ * FIN_ * 2);
  _Float16* Wl2h  = (_Float16*)carve((size_t)HID_ * HC_ * 2);
  _Float16* Wr2h  = (_Float16*)carve((size_t)HID_ * HC_ * 2);
  _Float16* xt16  = (_Float16*)carve((size_t)N_ * FIN_ * 2);
  float*    xl1   = (float*)carve((size_t)N_ * HC_ * 4);
  float*    xr1   = (float*)carve((size_t)N_ * HC_ * 4);
  float*    acc1  = (float*)carve((size_t)N_ * HC_ * 4);
  _Float16* h16   = (_Float16*)carve((size_t)N_ * HC_ * 2);
  float*    xl2   = (float*)carve((size_t)N_ * HID_ * 4);
  float*    xr2   = (float*)carve((size_t)N_ * HID_ * 4);
  float*    acc2  = (float*)carve((size_t)N_ * HID_ * 4);
  float*    lg1   = (float*)carve((size_t)EN_ * HEADS_ * 4);
  unsigned* mx1   = (unsigned*)carve((size_t)N_ * HEADS_ * 4);
  float*    dn1   = (float*)carve((size_t)N_ * HEADS_ * 4);
  float*    lg2   = (float*)carve((size_t)EN_ * 4);
  unsigned* mx2   = (unsigned*)carve((size_t)N_ * 4);
  float*    dn2   = (float*)carve((size_t)N_ * 4);
  float*    seqb  = (float*)carve((size_t)N_ * T_ * HID_ * 4);

  const int thr = 256;
  auto blocks = [](int n, int t) { return (n + t - 1) / t; };
  auto gemm_grid = [](int M, int Nout) {
    int waves = ((M + 15) / 16) * (Nout / 64);    // one wave per 16x64 strip
    return (waves + 7) / 8;                       // 8 waves per 256-thread block
  };
  const unsigned NEG_INF_ENC = 0x007FFFFFu;       // fenc(-inf)

  // edge-attr mean + extended edge attributes (self loops get the mean)
  fill_k<<<1, 64, 0, stream>>>((unsigned*)fsum, 1, 0u);
  edge_sum_k<<<256, thr, 0, stream>>>(eattr, fsum);
  ea_fill_k<<<blocks(EN_, thr), thr, 0, stream>>>(eattr, fsum, ea);

  // weights -> f16 (once per launch; deterministic)
  f2h_k<<<blocks(HC_ * FIN_, thr), thr, 0, stream>>>(Wl1, Wl1h, HC_ * FIN_);
  f2h_k<<<blocks(HC_ * FIN_, thr), thr, 0, stream>>>(Wr1, Wr1h, HC_ * FIN_);
  f2h_k<<<blocks(HID_ * HC_, thr), thr, 0, stream>>>(Wl2, Wl2h, HID_ * HC_);
  f2h_k<<<blocks(HID_ * HC_, thr), thr, 0, stream>>>(Wr2, Wr2h, HID_ * HC_);

  for (int t = 0; t < T_; ++t) {
    conv_xt_k<<<blocks(N_ * FIN_, thr), thr, 0, stream>>>(x, xt16, t);

    // layer-1 projections (WMMA)
    gemm_wmma_k<<<gemm_grid(N_, HC_), thr, 0, stream>>>(xt16, Wl1h, bl1, xl1, N_, FIN_, HC_);
    gemm_wmma_k<<<gemm_grid(N_, HC_), thr, 0, stream>>>(xt16, Wr1h, br1, xr1, N_, FIN_, HC_);

    fill_k<<<blocks(N_ * HEADS_, thr), thr, 0, stream>>>(mx1, N_ * HEADS_, NEG_INF_ENC);
    fill_k<<<blocks(N_ * HEADS_, thr), thr, 0, stream>>>((unsigned*)dn1, N_ * HEADS_, 0u);
    fill_k<<<blocks(N_ * HC_, thr), thr, 0, stream>>>((unsigned*)acc1, N_ * HC_, 0u);

    gat1_logits_k<<<blocks(EN_ * 32, thr), thr, 0, stream>>>(xl1, xr1, ea, ei, We1, att1, lg1, mx1);
    gat1_soft_k<<<blocks(EN_ * HEADS_, thr), thr, 0, stream>>>(ei, mx1, lg1, dn1);
    gat1_aggr_k<<<blocks(EN_ * 32, thr), thr, 0, stream>>>(xl1, ei, lg1, dn1, acc1);

    elu_h16_k<<<blocks(N_ * HC_, thr), thr, 0, stream>>>(acc1, bias1, h16);

    // layer-2 projections (WMMA)
    gemm_wmma_k<<<gemm_grid(N_, HID_), thr, 0, stream>>>(h16, Wl2h, bl2, xl2, N_, HC_, HID_);
    gemm_wmma_k<<<gemm_grid(N_, HID_), thr, 0, stream>>>(h16, Wr2h, br2, xr2, N_, HC_, HID_);

    fill_k<<<blocks(N_, thr), thr, 0, stream>>>(mx2, N_, NEG_INF_ENC);
    fill_k<<<blocks(N_, thr), thr, 0, stream>>>((unsigned*)dn2, N_, 0u);
    fill_k<<<blocks(N_ * HID_, thr), thr, 0, stream>>>((unsigned*)acc2, N_ * HID_, 0u);

    gat2_logits_k<<<blocks(EN_ * 32, thr), thr, 0, stream>>>(xl2, xr2, ea, ei, We2, att2, lg2, mx2);
    gat2_soft_k<<<blocks(EN_, thr), thr, 0, stream>>>(ei, mx2, lg2, dn2);
    gat2_aggr_k<<<blocks(EN_ * 32, thr), thr, 0, stream>>>(xl2, ei, lg2, dn2, acc2);

    seq_write_k<<<blocks(N_ * HID_, thr), thr, 0, stream>>>(acc2, bias2, seqb, t);
  }

  // fused temporal attention + layernorm + MLP head (wave per node)
  temporal_k<<<blocks(N_, 8), thr, 0, stream>>>(seqb, Wqkv, bqkv, Wo, bo, ln_g, ln_b,
                                                Wf1, bf1, Wout, bout, out);
}
